// TransformerBlock_43963285242070
// MI455X (gfx1250) — matrix operand order repaired and checked by reading, not compiled, not run on hardware
//
#include <hip/hip_runtime.h>
#include <hip/hip_bf16.h>

// ---------------------------------------------------------------------------
// Types / helpers
// ---------------------------------------------------------------------------
typedef __attribute__((ext_vector_type(16))) __bf16 v16bf;
typedef __attribute__((ext_vector_type(8)))  float  v8f;
typedef unsigned int  uint32;

union Frag16 {                     // one WMMA 16-bit operand (8 VGPRs)
    v16bf    v;
    uint32   u[8];
    unsigned short s[16];
};

struct MaskT   { static constexpr bool value = true;  };
struct NoMaskT { static constexpr bool value = false; };

__device__ __forceinline__ unsigned short f2bf(float f) {
    uint32 u = __float_as_uint(f);
    u += 0x7fffu + ((u >> 16) & 1u);           // round-to-nearest-even
    return (unsigned short)(u >> 16);
}
// element-pair -> offset inside a 16x32 A fragment row (in bf16 units)
__device__ __forceinline__ constexpr int ofsA(int v) {
    return 2 * v;                              // + h*16 supplied by base ptr
}

#define WMMA_BF16(A_, B_, C_) \
    __builtin_amdgcn_wmma_f32_16x16x32_bf16(false, (A_), false, (B_), (short)0, (C_), false, false)

// ---- CDNA5 async global->LDS path (probe-verified present) ----------------
#if __has_builtin(__builtin_amdgcn_global_load_async_to_lds_b128)
#define HAS_ASYNC_LDS 1
typedef int v4i_vs __attribute__((vector_size(4 * sizeof(int))));
typedef __attribute__((address_space(1))) v4i_vs* gv4p;   // global int4*
typedef __attribute__((address_space(3))) v4i_vs* lv4p;   // LDS int4*
#else
#define HAS_ASYNC_LDS 0
#endif

#if __has_builtin(__builtin_amdgcn_s_wait_asynccnt)
#define WAIT_ASYNC0() __builtin_amdgcn_s_wait_asynccnt(0)
#else
#define WAIT_ASYNC0() asm volatile("s_wait_asynccnt 0" ::: "memory")
#endif

// ---------------------------------------------------------------------------
// f32 -> bf16 weight conversion
// ---------------------------------------------------------------------------
__global__ __launch_bounds__(256) void cvt_bf16_kernel(const float* __restrict__ src,
                                                       unsigned short* __restrict__ dst,
                                                       int n) {
    int i = blockIdx.x * 256 + threadIdx.x;
    if (i < n) dst[i] = f2bf(src[i]);
}

// ---------------------------------------------------------------------------
// LayerNorm (D = 1024), one block per row, writes bf16
// ---------------------------------------------------------------------------
__global__ __launch_bounds__(256) void ln_bf16_kernel(const float* __restrict__ x,
                                                      const float* __restrict__ g,
                                                      const float* __restrict__ b,
                                                      unsigned short* __restrict__ out) {
    __shared__ float red[256];
    const int tid = threadIdx.x;
    const size_t row = blockIdx.x;
    float4 v = ((const float4*)(x + row * 1024))[tid];
    red[tid] = v.x + v.y + v.z + v.w;
    __syncthreads();
    for (int off = 128; off > 0; off >>= 1) {
        if (tid < off) red[tid] += red[tid + off];
        __syncthreads();
    }
    float mean = red[0] * (1.0f / 1024.0f);
    __syncthreads();
    float dx = v.x - mean, dy = v.y - mean, dz = v.z - mean, dw = v.w - mean;
    red[tid] = dx * dx + dy * dy + dz * dz + dw * dw;
    __syncthreads();
    for (int off = 128; off > 0; off >>= 1) {
        if (tid < off) red[tid] += red[tid + off];
        __syncthreads();
    }
    float rstd = rsqrtf(red[0] * (1.0f / 1024.0f) + 1e-5f);
    int c = tid * 4;
    out[row * 1024 + c + 0] = f2bf(dx * rstd * g[c + 0] + b[c + 0]);
    out[row * 1024 + c + 1] = f2bf(dy * rstd * g[c + 1] + b[c + 1]);
    out[row * 1024 + c + 2] = f2bf(dz * rstd * g[c + 2] + b[c + 2]);
    out[row * 1024 + c + 3] = f2bf(dw * rstd * g[c + 3] + b[c + 3]);
}

// ---------------------------------------------------------------------------
// Tiled bf16 WMMA GEMM, software-pipelined with double-buffered LDS.
// Block 256 thr (8 waves), tile 128x128x32; each wave: 64x32 (4x2 wmma frags)
// A tile: async global->LDS (uint pairs [128][20], padded).
// B tile: k-pair interleaved [16][132] packed in registers.
// ---------------------------------------------------------------------------
#define AW 20     // uint words per A row (16 data + 4 pad)
#define BW 132    // uint words per packed-B k-pair row (128 data + 4 pad)

template <bool GELU, bool RESID, bool OUT_BF16>
__global__ __launch_bounds__(256) void gemm_bf16_kernel(
        const unsigned short* __restrict__ A,    // [M,K] bf16
        const unsigned short* __restrict__ W,    // [K,N] bf16
        const float* __restrict__ bias,          // [N]
        const float* __restrict__ resid,         // [M,N] or null
        void* __restrict__ outp,                 // f32 or bf16 [M,N]
        int M, int N, int K) {
    __shared__ uint32 sA[2][128 * AW];           // 2 x 10240 B
    __shared__ uint32 sB[2][16 * BW];            // 2 x  8448 B
    const int tid = threadIdx.x;
    const int wave = tid >> 5, lane = tid & 31;
    const int h = lane >> 4, ml = lane & 15;
    const int m_blk = blockIdx.y * 128, n_blk = blockIdx.x * 128;
    const int wm0 = (wave >> 2) * 64;            // wave M offset in tile
    const int wn0 = (wave & 3) * 32;             // wave N offset in tile

    v8f zf = {};
    v8f acc[4][2];
#pragma unroll
    for (int mt = 0; mt < 4; ++mt)
#pragma unroll
        for (int nt = 0; nt < 2; ++nt) acc[mt][nt] = zf;

    // loader indices (fixed per thread)
    const int ra0 = (tid * 2) >> 2, ca0 = ((tid * 2) & 3) * 4;       // A chunk 0
    const int ra1 = (tid * 2 + 1) >> 2, ca1 = ((tid * 2 + 1) & 3) * 4;
    const int kp = tid >> 4, nc = (tid & 15) * 8;                    // B pair-row chunk

    const unsigned short* gA0 = A + (size_t)(m_blk + ra0) * K + ca0 * 2;
    const unsigned short* gA1 = A + (size_t)(m_blk + ra1) * K + ca1 * 2;
    const unsigned short* gB0 = W + (size_t)(2 * kp) * N + n_blk + nc;
    const unsigned short* gB1 = W + (size_t)(2 * kp + 1) * N + n_blk + nc;

    auto issueA = [&](int k0, int buf) {
#if HAS_ASYNC_LDS
        __builtin_amdgcn_global_load_async_to_lds_b128(
            (gv4p)(gA0 + k0), (lv4p)(&sA[buf][ra0 * AW + ca0]), 0, 0);
        __builtin_amdgcn_global_load_async_to_lds_b128(
            (gv4p)(gA1 + k0), (lv4p)(&sA[buf][ra1 * AW + ca1]), 0, 0);
#else
        *(uint4*)(&sA[buf][ra0 * AW + ca0]) = *(const uint4*)(gA0 + k0);
        *(uint4*)(&sA[buf][ra1 * AW + ca1]) = *(const uint4*)(gA1 + k0);
#endif
    };
    auto packB = [&](int buf, uint4 a4, uint4 b4) {
        const uint32* aw = (const uint32*)&a4;
        const uint32* bw = (const uint32*)&b4;
        uint32 o[8];
#pragma unroll
        for (int i = 0; i < 4; ++i) {
            o[2 * i]     = (aw[i] & 0xffffu) | (bw[i] << 16);
            o[2 * i + 1] = (aw[i] >> 16) | (bw[i] & 0xffff0000u);
        }
        uint4 p0; p0.x = o[0]; p0.y = o[1]; p0.z = o[2]; p0.w = o[3];
        uint4 p1; p1.x = o[4]; p1.y = o[5]; p1.z = o[6]; p1.w = o[7];
        *(uint4*)(&sB[buf][kp * BW + nc])     = p0;
        *(uint4*)(&sB[buf][kp * BW + nc + 4]) = p1;
    };

    const int nk = K >> 5;
    // ---- prologue: stage tile 0 into buffer 0 ----
    issueA(0, 0);
    packB(0, *(const uint4*)(gB0), *(const uint4*)(gB1));
    WAIT_ASYNC0();
    __syncthreads();

    for (int i = 0; i < nk; ++i) {
        const int cb = i & 1, nb = cb ^ 1;
        const bool hasNext = (i + 1 < nk);
        uint4 na4, nb4;
        if (hasNext) {                            // prefetch next tile
            const size_t koff = (size_t)(i + 1) << 5;
            issueA((int)koff, nb);
            na4 = *(const uint4*)(gB0 + koff * N);
            nb4 = *(const uint4*)(gB1 + koff * N);
        }
        // ---- fragments from current buffer: straight b32 LDS loads ----
        Frag16 bfrag[2];
        {
            const uint32* bb = &sB[cb][h * 8 * BW + wn0 + ml];
#pragma unroll
            for (int nt = 0; nt < 2; ++nt)
#pragma unroll
                for (int v = 0; v < 8; ++v)
                    bfrag[nt].u[v] = bb[v * BW + nt * 16];
        }
        Frag16 afrag[4];
#pragma unroll
        for (int mt = 0; mt < 4; ++mt) {
            const uint32* ab = &sA[cb][(wm0 + mt * 16 + ml) * AW + h * 8];
#pragma unroll
            for (int v = 0; v < 8; ++v)
                afrag[mt].u[v] = ab[v];
        }
#pragma unroll
        for (int mt = 0; mt < 4; ++mt)
#pragma unroll
            for (int nt = 0; nt < 2; ++nt)
                acc[mt][nt] = WMMA_BF16(afrag[mt].v, bfrag[nt].v, acc[mt][nt]);
        if (hasNext) packB(nb, na4, nb4);
        WAIT_ASYNC0();
        __syncthreads();
    }

    // epilogue
#pragma unroll
    for (int mt = 0; mt < 4; ++mt)
#pragma unroll
        for (int nt = 0; nt < 2; ++nt) {
            int n = n_blk + wn0 + nt * 16 + ml;
            float bv = bias[n];
#pragma unroll
            for (int r = 0; r < 8; ++r) {
                int m = m_blk + wm0 + mt * 16 + r + 8 * h;
                float val = acc[mt][nt][r] + bv;
                if (RESID) val += resid[(size_t)m * N + n];
                if (GELU)  val = 0.5f * val * (1.0f + erff(val * 0.70710678118654752f));
                if (OUT_BF16)
                    ((unsigned short*)outp)[(size_t)m * N + n] = f2bf(val);
                else
                    ((float*)outp)[(size_t)m * N + n] = val;
            }
        }
}

// ---------------------------------------------------------------------------
// Repack QKV f32 [B,S,3,H,hd] -> per-head bf16 [B,H,S,hd]; Q pre-scaled
// ---------------------------------------------------------------------------
__global__ __launch_bounds__(256) void repack_qkv_kernel(const float* __restrict__ qkv,
                                                         unsigned short* __restrict__ q,
                                                         unsigned short* __restrict__ k,
                                                         unsigned short* __restrict__ v) {
    size_t idx = (size_t)blockIdx.x * 256 + threadIdx.x;   // over 4096*3072
    int col = (int)(idx % 3072);
    size_t rowi = idx / 3072;                              // b*S + s
    int c = col >> 10;               // 0=q,1=k,2=v
    int rem = col & 1023;
    int hh = rem >> 6, dd = rem & 63;
    int b = (int)(rowi >> 11), s = (int)(rowi & 2047);
    float val = qkv[idx];
    size_t o = (((size_t)(b * 16 + hh) * 2048) + s) * 64 + dd;
    if (c == 0)      q[o] = f2bf(val * 0.125f);            // hd^-0.5 = 1/8
    else if (c == 1) k[o] = f2bf(val);
    else             v[o] = f2bf(val);
}

// ---------------------------------------------------------------------------
// Flash attention (causal), bf16 WMMA, f32 online softmax.
// Block 128 thr (4 waves); wave w owns q rows [q0, q0+16). KV step = 64.
// Unmasked steady-state blocks + one masked diagonal block.
// Row sums computed with WMMA against an all-ones fragment (no shuffles).
// ---------------------------------------------------------------------------
__global__ __launch_bounds__(128) void flash_attn_kernel(
        const unsigned short* __restrict__ Qg,   // [B,H,S,hd] bf16 (pre-scaled)
        const unsigned short* __restrict__ Kg,
        const unsigned short* __restrict__ Vg,
        unsigned short* __restrict__ O) {        // [B,S,D] bf16
    __shared__ unsigned short sP[4][16 * 64];    // per-wave P transpose (8 KB)
    const int S = 2048, HD = 64, D = 1024;
    const int wave = threadIdx.x >> 5, lane = threadIdx.x & 31;
    const int h = lane >> 4, ml = lane & 15;
    const int bh = blockIdx.x;
    const int b = bh >> 4, hh = bh & 15;
    const int q0 = blockIdx.y * 64 + wave * 16;

    // Q A-fragments (feature dim 64 -> two 16x32 frags), base + const offsets
    Frag16 qf[2];
    {
        const unsigned short* qb = Qg + (size_t)bh * S * HD + (size_t)(q0 + ml) * HD + h * 16;
#pragma unroll
        for (int c = 0; c < 2; ++c)
#pragma unroll
            for (int v = 0; v < 8; ++v)
                qf[c].u[v] = *(const uint32*)(qb + c * 32 + ofsA(v));
    }
    // constant all-ones B fragment (bf16 1.0 = 0x3F80)
    Frag16 ones;
#pragma unroll
    for (int v = 0; v < 8; ++v) ones.u[v] = 0x3F803F80u;

    v8f zf = {};
    v8f oa[4];
#pragma unroll
    for (int nt = 0; nt < 4; ++nt) oa[nt] = zf;
    float mrun[8], lrun[8];
    int qr[8];
#pragma unroll
    for (int r = 0; r < 8; ++r) {
        mrun[r] = -3.0e38f; lrun[r] = 0.0f;
        qr[r] = q0 + r + 8 * h;                  // causal row limit (invariant)
    }

    // per-lane base pointers, bumped by 64 rows (=4096 elems) per kv step
    const unsigned short* kb = Kg + (size_t)bh * S * HD + (size_t)ml * HD + h * 16;
    const unsigned short* vb = Vg + (size_t)bh * S * HD + (size_t)(h * 16) * HD + ml;
    unsigned short*       sps = &sP[wave][h * 8 * 64 + ml];       // store base
    const unsigned short* spl = &sP[wave][ml * 64 + h * 16];      // load base

    auto kv_block = [&](int j, auto masked_tag) {
        constexpr bool MASKED = decltype(masked_tag)::value;
        // ---- scores: 16x64 (four 16x16 C frags) ----
        v8f s[4];
#pragma unroll
        for (int c2 = 0; c2 < 4; ++c2) s[c2] = zf;
#pragma unroll
        for (int c2 = 0; c2 < 4; ++c2)
#pragma unroll
            for (int c = 0; c < 2; ++c) {
                Frag16 kf;
#pragma unroll
                for (int v = 0; v < 8; ++v)
                    kf.u[v] = *(const uint32*)(kb + c2 * 16 * HD + c * 32 + 2 * v);
                s[c2] = WMMA_BF16(qf[c].v, kf.v, s[c2]);
            }
        // ---- (optional) causal mask + online max (row = r+8h, col = lane&15)
        const int jml = j + ml;
        float scal[8];
#pragma unroll
        for (int r = 0; r < 8; ++r) {
            float v0 = s[0][r], v1 = s[1][r], v2 = s[2][r], v3 = s[3][r];
            if (MASKED) {
                v0 = (jml      <= qr[r]) ? v0 : -3.0e38f;
                v1 = (jml + 16 <= qr[r]) ? v1 : -3.0e38f;
                v2 = (jml + 32 <= qr[r]) ? v2 : -3.0e38f;
                v3 = (jml + 48 <= qr[r]) ? v3 : -3.0e38f;
            }
            float cm = fmaxf(fmaxf(v0, v1), fmaxf(v2, v3));
#pragma unroll
            for (int msk = 8; msk >= 1; msk >>= 1)
                cm = fmaxf(cm, __shfl_xor(cm, msk, 32));
            float mnew = fmaxf(mrun[r], cm);
            scal[r] = __expf(mrun[r] - mnew);
            mrun[r] = mnew;
            sps[r * 64 +  0] = f2bf(__expf(v0 - mnew));
            sps[r * 64 + 16] = f2bf(__expf(v1 - mnew));
            sps[r * 64 + 32] = f2bf(__expf(v2 - mnew));
            sps[r * 64 + 48] = f2bf(__expf(v3 - mnew));
#pragma unroll
            for (int nt = 0; nt < 4; ++nt) oa[nt][r] *= scal[r];
        }
        asm volatile("s_wait_dscnt 0" ::: "memory");   // wave-local LDS transpose
        // ---- P as two 16x32 A-fragments ----
        Frag16 pf[2];
#pragma unroll
        for (int c = 0; c < 2; ++c)
#pragma unroll
            for (int v = 0; v < 8; ++v)
                pf[c].u[v] = *(const uint32*)(spl + c * 32 + ofsA(v));
        // ---- row sums of P via ones-matrix WMMA (replicated over lanes) ----
        v8f sum = zf;
        sum = WMMA_BF16(pf[0].v, ones.v, sum);
        sum = WMMA_BF16(pf[1].v, ones.v, sum);
#pragma unroll
        for (int r = 0; r < 8; ++r) lrun[r] = lrun[r] * scal[r] + sum[r];
        // ---- O += P @ V   (2 kv chunks x 4 feature chunks) ----
#pragma unroll
        for (int c = 0; c < 2; ++c)
#pragma unroll
            for (int nt = 0; nt < 4; ++nt) {
                Frag16 vf;
#pragma unroll
                for (int v = 0; v < 8; ++v) {
                    vf.s[2 * v]     = vb[c * 32 * HD + v * 2 * HD + nt * 16];
                    vf.s[2 * v + 1] = vb[c * 32 * HD + v * 2 * HD + HD + nt * 16];
                }
                oa[nt] = WMMA_BF16(pf[c].v, vf.v, oa[nt]);
            }
        kb += 64 * HD;
        vb += 64 * HD;
    };

    const int jsplit = q0 & ~63;                 // first block touching diagonal
    for (int j = 0; j < jsplit; j += 64) kv_block(j, NoMaskT{});
    kv_block(jsplit, MaskT{});                   // exactly one masked block

    // ---- normalize + write [B,S,D] bf16 ----
    unsigned short* ob = O + ((size_t)(b * S + q0 + 8 * h)) * D + hh * HD + ml;
#pragma unroll
    for (int r = 0; r < 8; ++r) {
        float rl = __builtin_amdgcn_rcpf(lrun[r]);
#pragma unroll
        for (int nt = 0; nt < 4; ++nt)
            ob[r * D + nt * 16] = f2bf(oa[nt][r] * rl);
    }
}

// ---------------------------------------------------------------------------
// Host launcher
// ---------------------------------------------------------------------------
extern "C" void kernel_launch(void* const* d_in, const int* in_sizes, int n_in,
                              void* d_out, int out_size, void* d_ws, size_t ws_size,
                              hipStream_t stream) {
    (void)in_sizes; (void)n_in; (void)out_size; (void)ws_size;
    const int B = 2, S = 2048, D = 1024, F = 4096;
    const int M = B * S;                          // 4096

    const float* x     = (const float*)d_in[0];
    const float* ln1_g = (const float*)d_in[2];
    const float* ln1_b = (const float*)d_in[3];
    const float* Wqkv  = (const float*)d_in[4];
    const float* bqkv  = (const float*)d_in[5];
    const float* Wo    = (const float*)d_in[6];
    const float* bo    = (const float*)d_in[7];
    const float* ln2_g = (const float*)d_in[8];
    const float* ln2_b = (const float*)d_in[9];
    const float* W1    = (const float*)d_in[10];
    const float* b1    = (const float*)d_in[11];
    const float* W2    = (const float*)d_in[12];
    const float* b2    = (const float*)d_in[13];

    // workspace layout (bytes, all 16B aligned)
    char* w = (char*)d_ws;
    unsigned short* Wqkv_h = (unsigned short*)w; w += (size_t)D * 3 * D * 2;   // 6 MB
    unsigned short* Wo_h   = (unsigned short*)w; w += (size_t)D * D * 2;       // 2 MB
    unsigned short* W1_h   = (unsigned short*)w; w += (size_t)D * F * 2;       // 8 MB
    unsigned short* W2_h   = (unsigned short*)w; w += (size_t)F * D * 2;       // 8 MB
    unsigned short* h_bf   = (unsigned short*)w; w += (size_t)M * D * 2;       // 8 MB
    unsigned short* q_bf   = (unsigned short*)w; w += (size_t)M * D * 2;       // 8 MB
    unsigned short* k_bf   = (unsigned short*)w; w += (size_t)M * D * 2;       // 8 MB
    unsigned short* v_bf   = (unsigned short*)w; w += (size_t)M * D * 2;       // 8 MB
    unsigned short* at_bf  = (unsigned short*)w; w += (size_t)M * D * 2;       // 8 MB
    float*          y1     = (float*)w;          w += (size_t)M * D * 4;       // 16 MB
    unsigned short* g_bf   = (unsigned short*)w; w += (size_t)M * F * 2;       // 32 MB
    float*          qkv_f  = (float*)w;          w += (size_t)M * 3 * D * 4;   // 48 MB

    // 1) weights -> bf16
    cvt_bf16_kernel<<<(D * 3 * D) / 256, 256, 0, stream>>>(Wqkv, Wqkv_h, D * 3 * D);
    cvt_bf16_kernel<<<(D * D) / 256,     256, 0, stream>>>(Wo,   Wo_h,   D * D);
    cvt_bf16_kernel<<<(D * F) / 256,     256, 0, stream>>>(W1,   W1_h,   D * F);
    cvt_bf16_kernel<<<(F * D) / 256,     256, 0, stream>>>(W2,   W2_h,   F * D);

    // 2) LN1
    ln_bf16_kernel<<<M, 256, 0, stream>>>(x, ln1_g, ln1_b, h_bf);

    // 3) QKV = h @ Wqkv + bqkv   (f32 out)
    gemm_bf16_kernel<false, false, false>
        <<<dim3(3 * D / 128, M / 128), 256, 0, stream>>>(h_bf, Wqkv_h, bqkv, nullptr,
                                                         (void*)qkv_f, M, 3 * D, D);
    // 4) repack to per-head bf16 (Q pre-scaled)
    repack_qkv_kernel<<<(M * 3 * D) / 256, 256, 0, stream>>>(qkv_f, q_bf, k_bf, v_bf);

    // 5) flash attention -> attn_out bf16 [B,S,D]
    flash_attn_kernel<<<dim3(B * 16, S / 64), 128, 0, stream>>>(q_bf, k_bf, v_bf, at_bf);

    // 6) y1 = x + attn @ Wo + bo   (f32)
    gemm_bf16_kernel<false, true, false>
        <<<dim3(D / 128, M / 128), 256, 0, stream>>>(at_bf, Wo_h, bo, x, (void*)y1, M, D, D);

    // 7) LN2
    ln_bf16_kernel<<<M, 256, 0, stream>>>(y1, ln2_g, ln2_b, h_bf);

    // 8) g = gelu(h2 @ W1 + b1)   (bf16)
    gemm_bf16_kernel<true, false, true>
        <<<dim3(F / 128, M / 128), 256, 0, stream>>>(h_bf, W1_h, b1, nullptr, (void*)g_bf, M, F, D);

    // 9) out = y1 + g @ W2 + b2   (f32 -> d_out)
    gemm_bf16_kernel<false, true, false>
        <<<dim3(D / 128, M / 128), 256, 0, stream>>>(g_bf, W2_h, b2, y1, d_out, M, D, F);
}
